// CausalSelfAttention_57509612093844
// MI455X (gfx1250) — compile-verified, matrix operand order checked by reading, not run on hardware
//
#include <hip/hip_runtime.h>
#include <hip/hip_bf16.h>

typedef _Float16 f16_t;
typedef __attribute__((ext_vector_type(16))) _Float16 v16h;
typedef __attribute__((ext_vector_type(8)))  _Float16 v8h;
typedef __attribute__((ext_vector_type(8)))  float    v8f;
typedef __attribute__((ext_vector_type(4)))  float    v4f;
typedef __attribute__((ext_vector_type(4)))  unsigned v4u;
typedef __attribute__((ext_vector_type(8)))  unsigned v8u;

#define NH    32
#define NKV   8
#define HD    128
#define SEQ   2048
#define BATCH 2
#define HID   4096
#define NQKV  6144   /* NH*HD + 2*NKV*HD */

// Build a 16x32-f16 A/B fragment for v_wmma_f32_16x16x32_f16 from two
// contiguous 8-halfword runs in LDS (K pattern: [kh..kh+7], [16+kh..16+kh+7]).
static __device__ __forceinline__ v16h make_frag(const f16_t* lo, const f16_t* hi) {
  v8h a = *(const v8h*)lo;
  v8h b = *(const v8h*)hi;
  v16h r;
#pragma unroll
  for (int i = 0; i < 8; ++i) { r[i] = a[i]; r[8 + i] = b[i]; }
  return r;
}

// LDS byte offset of a generic pointer into __shared__ (low 32 bits of the
// flat address are the wave-relative LDS offset on gfx1250).
static __device__ __forceinline__ unsigned lds_off(const void* p) {
  return (unsigned)(uintptr_t)p;
}

// Async global -> LDS 16-byte copy (ASYNCcnt-tracked, no VGPR round trip).
static __device__ __forceinline__ void async_copy_b128(void* lptr, const void* gptr) {
  unsigned loff = lds_off(lptr);
  unsigned long long g = (unsigned long long)(uintptr_t)gptr;
  asm volatile("global_load_async_to_lds_b128 %0, %1, off"
               :: "v"(loff), "v"(g) : "memory");
}
static __device__ __forceinline__ void wait_async0() {
  asm volatile("s_wait_asynccnt 0" ::: "memory");
}

// Tensor Data Mover: 2D tile load global -> LDS with optional row padding.
// rows x cols elements of 2 bytes, global row stride = stride elems.
// pad_interval/pad_amount use the encoded D# fields (ISA 8.4).
static __device__ __forceinline__ void tdm_load_2d(unsigned lds_addr, const void* gptr,
                                                   unsigned rows, unsigned cols,
                                                   unsigned stride,
                                                   unsigned pad_en,
                                                   unsigned pad_interval,
                                                   unsigned pad_amount) {
  unsigned long long ga = (unsigned long long)(uintptr_t)gptr;
  v4u g0;
  g0[0] = 1u;                                                   // count=1, user mode
  g0[1] = lds_addr;                                             // lds_addr [63:32]
  g0[2] = (unsigned)ga;                                         // global_addr lo
  g0[3] = (unsigned)((ga >> 32) & 0x01FFFFFFu) | (2u << 30);    // addr hi | type=2
  v8u g1;
  g1[0] = (1u << 16) | (pad_en << 20) | (pad_interval << 22) | (pad_amount << 25);
  g1[1] = (cols & 0xFFFFu) << 16;                               // tensor_dim0[15:0]
  g1[2] = ((cols >> 16) & 0xFFFFu) | ((rows & 0xFFFFu) << 16);  // dim0 hi | tensor_dim1 lo
  g1[3] = ((rows >> 16) & 0xFFFFu) | ((cols & 0xFFFFu) << 16);  // dim1 hi | tile_dim0
  g1[4] = rows & 0xFFFFu;                                       // tile_dim1 (tile_dim2=0)
  g1[5] = stride;                                               // tensor_dim0_stride lo32
  g1[6] = 0u;                                                   // stride hi | dim1_stride lo
  g1[7] = 0u;
  asm volatile("tensor_load_to_lds %0, %1" :: "s"(g0), "s"(g1) : "memory");
}

// ---------------------------------------------------------------------------
// Generic f16-WMMA GEMM: C(MxN) = A(MxK) * B(KxN), f32 accumulate.
// A may be f32 (Af, converted during staging) or f16 (Ah, async-copied);
// B is f32; out is f16 (Ch) or f32 (Cf).
// Block: 256 threads = 8 waves; block tile 128x128; wave tile 64x32.
// ---------------------------------------------------------------------------
__global__ __launch_bounds__(256)
void gemm_f16wmma(const float* __restrict__ Af, const f16_t* __restrict__ Ah,
                  const float* __restrict__ Bf,
                  f16_t* __restrict__ Ch, float* __restrict__ Cf,
                  int M, int N, int K) {
  __shared__ f16_t As[128][40];   // [m][k], padded stride 40 halves (80B)
  __shared__ f16_t Bs[128][40];   // transposed: [n][k]

  const int tid  = threadIdx.x;
  const int lane = tid & 31;
  const int wid  = tid >> 5;          // 0..7
  const int wm   = (wid & 1) * 64;    // wave row offset in block tile
  const int wn   = (wid >> 1) * 32;   // wave col offset in block tile
  const int hf   = lane >> 4;         // half-wave id
  const int l16  = lane & 15;
  const int kh   = hf * 8;            // per-half K offset in fragments
  const int bm   = blockIdx.y * 128;
  const int bn   = blockIdx.x * 128;

  v8f acc[4][2];
#pragma unroll
  for (int mt = 0; mt < 4; ++mt)
#pragma unroll
    for (int nt = 0; nt < 2; ++nt) acc[mt][nt] = (v8f){};

  for (int kk = 0; kk < K; kk += 32) {
    __syncthreads();
    // ---- stage A tile (128 x 32) into LDS as f16 ----
    if (Ah) {
#pragma unroll
      for (int i = 0; i < 2; ++i) {
        int slot = tid + i * 256;          // 512 slots x 8 halves
        int m = slot >> 2, c = (slot & 3) * 8;
        async_copy_b128(&As[m][c], Ah + (size_t)(bm + m) * K + kk + c);
      }
    } else {
#pragma unroll
      for (int i = 0; i < 4; ++i) {
        int slot = tid + i * 256;          // 1024 slots x 4 floats
        int m = slot >> 3, c = (slot & 7) * 4;
        v4f t = *(const v4f*)(Af + (size_t)(bm + m) * K + kk + c);
#pragma unroll
        for (int j = 0; j < 4; ++j) As[m][c + j] = (f16_t)t[j];
      }
    }
    // ---- stage B tile (32 x 128) transposed into LDS as f16 ----
#pragma unroll
    for (int i = 0; i < 4; ++i) {
      int slot = tid + i * 256;            // 1024 slots x 4 floats
      int k = slot >> 5, n0 = (slot & 31) * 4;
      v4f t = *(const v4f*)(Bf + (size_t)(kk + k) * N + bn + n0);
#pragma unroll
      for (int j = 0; j < 4; ++j) Bs[n0 + j][k] = (f16_t)t[j];
    }
    // ---- prefetch next K-step tiles into L2 while we compute ----
    if (kk + 32 < K) {
      int mP = tid >> 3, cP = (tid & 7) * 4;
      if (!Ah) __builtin_prefetch(Af + (size_t)(bm + mP) * K + kk + 32 + cP, 0, 1);
      else     __builtin_prefetch(Ah + (size_t)(bm + ((tid >> 2) & 127)) * K + kk + 32, 0, 1);
      int kP = tid >> 5, nP = (tid & 31) * 4;
      __builtin_prefetch(Bf + (size_t)(kk + 32 + kP) * N + bn + nP, 0, 1);
    }
    wait_async0();
    __syncthreads();

    v16h afr[4], bfr[2];
#pragma unroll
    for (int mt = 0; mt < 4; ++mt) {
      const f16_t* row = &As[wm + mt * 16 + l16][0];
      afr[mt] = make_frag(row + kh, row + 16 + kh);
    }
#pragma unroll
    for (int nt = 0; nt < 2; ++nt) {
      const f16_t* row = &Bs[wn + nt * 16 + l16][0];
      bfr[nt] = make_frag(row + kh, row + 16 + kh);
    }
#pragma unroll
    for (int mt = 0; mt < 4; ++mt)
#pragma unroll
      for (int nt = 0; nt < 2; ++nt)
        acc[mt][nt] = __builtin_amdgcn_wmma_f32_16x16x32_f16(
            false, afr[mt], false, bfr[nt], (short)0, acc[mt][nt], false, false);
  }

  // epilogue: C layout m = v + 8*hf, n = l16
#pragma unroll
  for (int mt = 0; mt < 4; ++mt)
#pragma unroll
    for (int nt = 0; nt < 2; ++nt)
#pragma unroll
      for (int v = 0; v < 8; ++v) {
        int gm = bm + wm + mt * 16 + v + 8 * hf;
        int gn = bn + wn + nt * 16 + l16;
        if (Ch) Ch[(size_t)gm * N + gn] = (f16_t)acc[mt][nt][v];
        else    Cf[(size_t)gm * N + gn] = acc[mt][nt][v];
      }
}

// ---------------------------------------------------------------------------
// RoPE on Q/K + split into head-major f16 Q[B,NH,S,HD], K/V[B,NKV,S,HD].
// ---------------------------------------------------------------------------
__global__ __launch_bounds__(256)
void rope_split(const f16_t* __restrict__ qkv,
                const float* __restrict__ cosT, const float* __restrict__ sinT,
                f16_t* __restrict__ Q, f16_t* __restrict__ Ko, f16_t* __restrict__ Vo) {
  size_t idx = (size_t)blockIdx.x * 256 + threadIdx.x;
  int row = (int)(idx / NQKV);
  int c   = (int)(idx % NQKV);
  int b = row >> 11, s = row & (SEQ - 1);
  float val = (float)qkv[idx];
  if (c < NH * HD) {
    int d = c & 127, h = c >> 7;
    float pv  = (float)qkv[(size_t)row * NQKV + (d < 64 ? c + 64 : c - 64)];
    float rot = d < 64 ? -pv : pv;
    float r = val * cosT[s * HD + d] + rot * sinT[s * HD + d];
    Q[(((size_t)(b * NH + h)) * SEQ + s) * HD + d] = (f16_t)r;
  } else if (c < NH * HD + NKV * HD) {
    int cc = c - NH * HD;
    int d = cc & 127, h = cc >> 7;
    float pv  = (float)qkv[(size_t)row * NQKV + NH * HD + (d < 64 ? cc + 64 : cc - 64)];
    float rot = d < 64 ? -pv : pv;
    float r = val * cosT[s * HD + d] + rot * sinT[s * HD + d];
    Ko[(((size_t)(b * NKV + h)) * SEQ + s) * HD + d] = (f16_t)r;
  } else {
    int cc = c - NH * HD - NKV * HD;
    int d = cc & 127, h = cc >> 7;
    Vo[(((size_t)(b * NKV + h)) * SEQ + s) * HD + d] = (f16_t)val;
  }
}

// ---------------------------------------------------------------------------
// Flash attention (causal, GQA 4:1). Block = (b,h) x 128 queries, 8 waves x
// 16 query rows each. Q tile and per-step K tile arrive via the Tensor Data
// Mover (padded rows in hardware); 32 keys per step: 8 WMMA Q*K^T, 8 WMMA P*V.
// ---------------------------------------------------------------------------
__global__ __launch_bounds__(256)
void flash_attn(const f16_t* __restrict__ Q, const f16_t* __restrict__ Kk,
                const f16_t* __restrict__ Vv, f16_t* __restrict__ Oh) {
  __shared__ f16_t Qs[128][136];     // queries x dim (pad 136 -> 272B rows)
  __shared__ f16_t Ks[32][136];      // keys x dim
  __shared__ f16_t Vt[128][40];      // dim x keys (transposed for B frags)
  __shared__ f16_t Ps[8][16][40];    // per-wave P tile (16 x 32, padded)

  const int tid  = threadIdx.x;
  const int lane = tid & 31;
  const int wid  = tid >> 5;
  const int hf   = lane >> 4;
  const int l16  = lane & 15;
  const int kh   = hf * 8;
  const int bh = blockIdx.x, b = bh >> 5, h = bh & 31;
  const int kvh = h >> 2;                       // 32/8 = 4 heads per KV head
  const int qBase = blockIdx.y * 128;
  const size_t qRow  = ((size_t)(b * NH + h)) * SEQ + qBase;
  const size_t kvRow = ((size_t)(b * NKV + kvh)) * SEQ;

  // ---- Q block (128 x 128 halves) via TDM: pad 4 DWORDs every 64 DWORDs ----
  if (wid == 0) {
    tdm_load_2d(lds_off(&Qs[0][0]), Q + qRow * HD,
                128 /*rows*/, 128 /*cols*/, 128 /*stride*/,
                1 /*pad_en*/, 5 /*64 DW*/, 3 /*4 DW*/);
    __builtin_amdgcn_s_wait_tensorcnt(0);
  }
  __syncthreads();

  float mi[8], li[8];
  v8f o[8];
#pragma unroll
  for (int v = 0; v < 8; ++v) { mi[v] = -1e30f; li[v] = 0.f; }
#pragma unroll
  for (int nt = 0; nt < 8; ++nt) o[nt] = (v8f){};

  const int   nkb  = (qBase + 128) >> 5;
  const int   qTop = qBase + wid * 16 + 15;     // last query row of this wave
  const float sc   = 0.08838834764831845f;      // 1/sqrt(128)

  for (int kb = 0; kb < nkb; ++kb) {
    const int kBase = kb * 32;
    __syncthreads();
    // ---- K tile (32 x 128) via TDM, V^T staged with manual transpose ----
    if (wid == 0)
      tdm_load_2d(lds_off(&Ks[0][0]), Kk + (kvRow + kBase) * HD,
                  32, 128, 128, 1, 5, 3);
#pragma unroll
    for (int i = 0; i < 2; ++i) {
      int slot = tid + i * 256;
      int key = slot >> 4, c8 = (slot & 15) * 8;
      v8h vv = *(const v8h*)(Vv + (kvRow + kBase + key) * HD + c8);
#pragma unroll
      for (int j = 0; j < 8; ++j) Vt[c8 + j][key] = vv[j];
    }
    if (kBase + 32 < nkb * 32) {   // prefetch next key block into L2
      int key = tid >> 4, c8 = (tid & 15) * 8;
      __builtin_prefetch(Kk + (kvRow + kBase + 32 + key) * HD + c8, 0, 1);
      __builtin_prefetch(Vv + (kvRow + kBase + 32 + key) * HD + c8, 0, 1);
    }
    if (wid == 0) __builtin_amdgcn_s_wait_tensorcnt(0);
    __syncthreads();
    if (kBase > qTop) continue;                 // wave fully above causal line

    // ---- S = Q * K^T  (16 x 32 per wave) ----
    v8f c0 = (v8f){}, c1 = (v8f){};
#pragma unroll
    for (int kt = 0; kt < 4; ++kt) {
      int dB = kt * 32;
      const f16_t* qr = &Qs[wid * 16 + l16][dB];
      v16h aq = make_frag(qr + kh, qr + 16 + kh);
      const f16_t* k0 = &Ks[l16][dB];
      const f16_t* k1 = &Ks[16 + l16][dB];
      v16h b0 = make_frag(k0 + kh, k0 + 16 + kh);
      v16h b1 = make_frag(k1 + kh, k1 + 16 + kh);
      c0 = __builtin_amdgcn_wmma_f32_16x16x32_f16(false, aq, false, b0, (short)0, c0, false, false);
      c1 = __builtin_amdgcn_wmma_f32_16x16x32_f16(false, aq, false, b1, (short)0, c1, false, false);
    }

    // ---- online softmax (rows m = v + 8*hf spread over 16 lanes) ----
#pragma unroll
    for (int v = 0; v < 8; ++v) {
      float s0 = c0[v] * sc, s1 = c1[v] * sc;
      int q = qBase + wid * 16 + v + 8 * hf;
      if (kBase + l16 > q)      s0 = -1e30f;
      if (kBase + 16 + l16 > q) s1 = -1e30f;
      float t = fmaxf(s0, s1);
#pragma unroll
      for (int off = 1; off < 16; off <<= 1) t = fmaxf(t, __shfl_xor(t, off, 32));
      float mn  = fmaxf(mi[v], t);
      float scl = __expf(mi[v] - mn);
      float p0  = __expf(s0 - mn);
      float p1  = __expf(s1 - mn);
      int m = v + 8 * hf;
      Ps[wid][m][l16]      = (f16_t)p0;
      Ps[wid][m][16 + l16] = (f16_t)p1;
      float rs = p0 + p1;
#pragma unroll
      for (int off = 1; off < 16; off <<= 1) rs += __shfl_xor(rs, off, 32);
      li[v] = li[v] * scl + rs;
      mi[v] = mn;
#pragma unroll
      for (int nt = 0; nt < 8; ++nt) o[nt][v] *= scl;
    }

    // ---- O += P * V  (16 x 128 per wave) ----
    const f16_t* pr = &Ps[wid][l16][0];
    v16h pa = make_frag(pr + kh, pr + 16 + kh);
#pragma unroll
    for (int nt = 0; nt < 8; ++nt) {
      const f16_t* vr = &Vt[nt * 16 + l16][0];
      v16h vb = make_frag(vr + kh, vr + 16 + kh);
      o[nt] = __builtin_amdgcn_wmma_f32_16x16x32_f16(false, pa, false, vb, (short)0, o[nt], false, false);
    }
  }

  // ---- normalize and store attn output as f16 [B*S, NH*HD] ----
#pragma unroll
  for (int nt = 0; nt < 8; ++nt)
#pragma unroll
    for (int v = 0; v < 8; ++v) {
      int m = v + 8 * hf;
      size_t srow = (size_t)b * SEQ + qBase + wid * 16 + m;
      int col = h * HD + nt * 16 + l16;
      Oh[srow * HID + col] = (f16_t)(o[nt][v] / li[v]);
    }
}

// ---------------------------------------------------------------------------
extern "C" void kernel_launch(void* const* d_in, const int* in_sizes, int n_in,
                              void* d_out, int out_size, void* d_ws, size_t ws_size,
                              hipStream_t stream) {
  (void)in_sizes; (void)n_in; (void)out_size; (void)ws_size;
  const float* x    = (const float*)d_in[0];
  const float* cosT = (const float*)d_in[1];
  const float* sinT = (const float*)d_in[2];
  const float* Wqkv = (const float*)d_in[3];
  const float* Wout = (const float*)d_in[4];
  float* out = (float*)d_out;

  char* ws = (char*)d_ws;
  const size_t qkv_bytes = (size_t)BATCH * SEQ * NQKV * sizeof(f16_t);        // 50,331,648
  const size_t q_bytes   = (size_t)BATCH * NH  * SEQ * HD * sizeof(f16_t);    // 33,554,432
  const size_t kv_bytes  = (size_t)BATCH * NKV * SEQ * HD * sizeof(f16_t);    //  8,388,608
  f16_t* qkvh  = (f16_t*)ws;
  f16_t* Qh    = (f16_t*)(ws + qkv_bytes);
  f16_t* Kh    = (f16_t*)(ws + qkv_bytes + q_bytes);
  f16_t* Vh    = (f16_t*)(ws + qkv_bytes + q_bytes + kv_bytes);
  f16_t* attnh = qkvh;   // qkv dead after rope_split; reuse for attn output

  dim3 blk(256);
  // 1) qkv = x @ Wqkv   (M=4096, N=6144, K=4096), f32 in, f16 out
  gemm_f16wmma<<<dim3(NQKV / 128, (BATCH * SEQ) / 128), blk, 0, stream>>>(
      x, nullptr, Wqkv, qkvh, nullptr, BATCH * SEQ, NQKV, HID);
  // 2) RoPE + head-major split
  size_t total = (size_t)BATCH * SEQ * NQKV;
  rope_split<<<dim3((unsigned)(total / 256)), blk, 0, stream>>>(
      qkvh, cosT, sinT, Qh, Kh, Vh);
  // 3) causal flash attention
  flash_attn<<<dim3(BATCH * NH, SEQ / 128), blk, 0, stream>>>(Qh, Kh, Vh, attnh);
  // 4) out = attn @ Wout (M=4096, N=4096, K=4096), f16 A, f32 out
  gemm_f16wmma<<<dim3(HID / 128, (BATCH * SEQ) / 128), blk, 0, stream>>>(
      nullptr, attnh, Wout, nullptr, out, BATCH * SEQ, HID, HID);
}
